// GCNModel_15470472200268
// MI455X (gfx1250) — matrix operand order repaired and checked by reading, not compiled, not run on hardware
//
#include <hip/hip_runtime.h>
#include <stdint.h>

#define IN_CH  128
#define HIDDEN 256

typedef __attribute__((ext_vector_type(16))) _Float16 v16h;
typedef __attribute__((ext_vector_type(8)))  float    v8f;
typedef __attribute__((ext_vector_type(4)))  float    v4f;

// ---------------------------------------------------------------- utilities
__global__ void zero_f32_kernel(float* __restrict__ p, size_t n) {
    size_t n4 = n >> 2;
    v4f* p4 = (v4f*)p;
    v4f zv = {0.0f, 0.0f, 0.0f, 0.0f};
    size_t i = (size_t)blockIdx.x * blockDim.x + threadIdx.x;
    size_t stride = (size_t)gridDim.x * blockDim.x;
    for (size_t j = i; j < n4; j += stride) p4[j] = zv;
    for (size_t j = (n4 << 2) + i; j < n; j += stride) p[j] = 0.0f;
}

// Pre-swizzle w1 (128x256 f32) into WMMA B-fragment-major f16 layout.
// Fragment id f = kc*16 + ct  (kc: k-chunk of 32, ct: 16-wide column tile).
// Within a fragment: lane l (col = ct*16 + (l&15), k-base = kc*32 + (l>>4)*16)
// holds 16 contiguous f16: element t = B[kbase + t][col].
__global__ void cvt_w1_swizzle_kernel(const float* __restrict__ w1,
                                      _Float16* __restrict__ w1f) {
    int idx = blockIdx.x * blockDim.x + threadIdx.x;   // exactly 32768 threads
    int f      = idx >> 9;          // 512 elements per fragment
    int within = idx & 511;
    int lane   = within >> 4;
    int t      = within & 15;
    int kc = f >> 4;
    int ct = f & 15;
    int col = ct * 16 + (lane & 15);
    int k   = kc * 32 + ((lane >> 4) * 16) + t;
    w1f[idx] = (_Float16)w1[(size_t)k * HIDDEN + col];
}

// in-degree (+1 for self-loop added later in dinv)
__global__ void deg_kernel(const int* __restrict__ dst, float* __restrict__ deg, int E) {
    int i = blockIdx.x * blockDim.x + threadIdx.x;
    if (i < E) atomicAdd(&deg[dst[i]], 1.0f);
}

__global__ void dinv_kernel(float* __restrict__ deg, int N) {
    int i = blockIdx.x * blockDim.x + threadIdx.x;
    if (i < N) {
        float d = deg[i] + 1.0f;   // self-loop
        deg[i] = rsqrtf(d);        // d >= 1 always
    }
}

// ---------------------------------------------------------------- WMMA GEMM
// h[N,256] = x[N,128] @ w1[128,256], f16 operands, f32 accumulate.
// One block per 16-row strip (N % 16 == 0). 8 wave32s; each wave owns two
// 16x16 output tiles (columns wave*32 .. wave*32+31), 4 k-chunks of 32.
// A fragments come from a 4 KB LDS strip (ds_load_b128 x2); B fragments are
// single 32-byte aligned vector loads from the pre-swizzled weight buffer.
__global__ void __launch_bounds__(256) gemm_x_w1_kernel(const float* __restrict__ x,
                                                        const _Float16* __restrict__ w1f,
                                                        float* __restrict__ h) {
    __shared__ _Float16 As[16 * IN_CH];          // 4 KB strip of x in f16
    const int r0 = blockIdx.x * 16;

    for (int idx = threadIdx.x; idx < 16 * IN_CH; idx += 256) {
        As[idx] = (_Float16)x[(size_t)(r0 + (idx >> 7)) * IN_CH + (idx & 127)];
    }
    __syncthreads();

    const int wave = threadIdx.x >> 5;
    const int lane = threadIdx.x & 31;
    const int row  = lane & 15;
    const int hi   = lane >> 4;          // 0: lanes 0-15, 1: lanes 16-31
    const int n0   = wave * 32;          // this wave covers columns [n0, n0+32)

    v8f acc0 = {};
    v8f acc1 = {};

    #pragma unroll
    for (int kc = 0; kc < 4; ++kc) {
        const int k0 = kc * 32;
        const int kb = hi * 16;
        // A fragment: contiguous k order (k = 16*hi + s), matching the B swizzle;
        // one 16-halfword run -> ds_load_b128 x2.
        v16h a;
        #pragma unroll
        for (int j = 0; j < 8; ++j) {
            a[j]     = As[row * IN_CH + k0 + kb + j];
            a[8 + j] = As[row * IN_CH + k0 + kb + 8 + j];
        }
        // B fragments: one aligned 32B vector load each.
        const v16h b0 = *(const v16h*)(w1f + ((size_t)(kc * 16 + 2 * wave)     << 9) + lane * 16);
        const v16h b1 = *(const v16h*)(w1f + ((size_t)(kc * 16 + 2 * wave + 1) << 9) + lane * 16);

        acc0 = __builtin_amdgcn_wmma_f32_16x16x32_f16(false, a, false, b0,
                                                      (short)0, acc0, false, false);
        acc1 = __builtin_amdgcn_wmma_f32_16x16x32_f16(false, a, false, b1,
                                                      (short)0, acc1, false, false);
    }

    // D layout: VGPR r -> lanes 0-15 (M=r, N=lane), lanes 16-31 (M=8+r, N=lane-16)
    #pragma unroll
    for (int r = 0; r < 8; ++r) {
        const int m = r + 8 * hi;
        const int n = lane & 15;
        h[(size_t)(r0 + m) * HIDDEN + n0 + n]      = acc0[r];
        h[(size_t)(r0 + m) * HIDDEN + n0 + 16 + n] = acc1[r];
    }
}

// ---------------------------------------------------------------- aggregation
// One wave32 per (edge or self-loop). Gather side: two non-temporal b128 loads
// per lane (512 B coalesced per wave-half); scatter side: f32 atomics that
// resolve in the 192 MB L2 (h and h2 are 51 MB each -> fully L2-resident).
__global__ void aggregate_kernel(const int* __restrict__ src, const int* __restrict__ dst,
                                 const float* __restrict__ dinv,
                                 const float* __restrict__ h, float* __restrict__ h2,
                                 int E, int N) {
    int gtid = blockIdx.x * blockDim.x + threadIdx.x;
    int w    = gtid >> 5;
    int lane = gtid & 31;
    if (w >= E + N) return;
    int s, d; float nrm;
    if (w < E) {
        s = src[w]; d = dst[w];
        nrm = dinv[s] * dinv[d];
    } else {
        s = d = w - E;
        float di = dinv[s];
        nrm = di * di;
    }
    const v4f* hs4 = (const v4f*)(h + (size_t)s * HIDDEN);   // rows are 16B aligned
    float*     hd  = h2 + (size_t)d * HIDDEN;
    #pragma unroll
    for (int half = 0; half < 2; ++half) {
        const int c = half * 128 + lane * 4;
        v4f v = __builtin_nontemporal_load(hs4 + half * 32 + lane);
        atomicAdd(&hd[c + 0], v.x * nrm);
        atomicAdd(&hd[c + 1], v.y * nrm);
        atomicAdd(&hd[c + 2], v.z * nrm);
        atomicAdd(&hd[c + 3], v.w * nrm);
    }
}

// bias + relu + global_mean_pool accumulation; one thread per 4 channels.
__global__ void bias_relu_pool_kernel(const float* __restrict__ h2, const float* __restrict__ b1,
                                      const int* __restrict__ batch,
                                      float* __restrict__ pooled, float* __restrict__ cnt,
                                      int N) {
    size_t idx = (size_t)blockIdx.x * blockDim.x + threadIdx.x;   // N * 64 threads
    if (idx >= (size_t)N * (HIDDEN / 4)) return;
    int i  = (int)(idx >> 6);
    int cq = (int)(idx & 63);
    int c  = cq * 4;
    v4f v = __builtin_nontemporal_load((const v4f*)h2 + idx);
    v4f b = *((const v4f*)b1 + cq);
    int g = batch[i];
    float* pg = pooled + (size_t)g * HIDDEN + c;
    atomicAdd(&pg[0], fmaxf(v.x + b.x, 0.0f));
    atomicAdd(&pg[1], fmaxf(v.y + b.y, 0.0f));
    atomicAdd(&pg[2], fmaxf(v.z + b.z, 0.0f));
    atomicAdd(&pg[3], fmaxf(v.w + b.w, 0.0f));
    if (cq == 0) atomicAdd(&cnt[g], 1.0f);
}

// z[g][j] = relu(b2[j] + sum_k (pooled[g][k]/cnt[g]) * w2[k][j])
__global__ void head1_kernel(const float* __restrict__ pooled, const float* __restrict__ cnt,
                             const float* __restrict__ w2, const float* __restrict__ b2,
                             float* __restrict__ z) {
    int g = blockIdx.x;
    int j = threadIdx.x;
    float invc = 1.0f / fmaxf(cnt[g], 1.0f);
    float acc = b2[j];
    const float* pg = pooled + (size_t)g * HIDDEN;
    #pragma unroll 4
    for (int k = 0; k < HIDDEN; ++k) {
        acc += (pg[k] * invc) * w2[(size_t)k * HIDDEN + j];
    }
    z[(size_t)g * HIDDEN + j] = fmaxf(acc, 0.0f);
}

// out[g] = sum_j z[g][j]*w3[j] + b3
__global__ void head2_kernel(const float* __restrict__ z, const float* __restrict__ w3,
                             const float* __restrict__ b3, float* __restrict__ out) {
    __shared__ float red[256];
    int g = blockIdx.x;
    int t = threadIdx.x;
    red[t] = z[(size_t)g * HIDDEN + t] * w3[t];
    __syncthreads();
    for (int s = 128; s > 0; s >>= 1) {
        if (t < s) red[t] += red[t + s];
        __syncthreads();
    }
    if (t == 0) out[g] = red[0] + b3[0];
}

// ---------------------------------------------------------------- launcher
extern "C" void kernel_launch(void* const* d_in, const int* in_sizes, int n_in,
                              void* d_out, int out_size, void* d_ws, size_t ws_size,
                              hipStream_t stream) {
    const float* x     = (const float*)d_in[0];
    const int*   ei    = (const int*)  d_in[1];   // [2, E] row-major
    const int*   batch = (const int*)  d_in[2];
    const float* w1    = (const float*)d_in[3];
    const float* b1    = (const float*)d_in[4];
    const float* w2    = (const float*)d_in[5];
    const float* b2    = (const float*)d_in[6];
    const float* w3    = (const float*)d_in[7];
    const float* b3    = (const float*)d_in[8];
    float* out = (float*)d_out;

    const int N = in_sizes[0] / IN_CH;   // 50000 (multiple of 16)
    const int E = in_sizes[1] / 2;       // 800000
    const int G = out_size;              // 64

    // workspace layout (floats)
    float* ws     = (float*)d_ws;
    float* h2     = ws;                                  // N*HIDDEN
    float* deg    = h2 + (size_t)N * HIDDEN;             // N  (becomes dinv)
    float* pooled = deg + N;                             // G*HIDDEN
    float* cnt    = pooled + (size_t)G * HIDDEN;         // G
    size_t zcount = (size_t)N * HIDDEN + N + (size_t)G * HIDDEN + G;
    float* hbuf   = cnt + G;                             // N*HIDDEN (16B-aligned: zcount%4==0)
    float* z      = hbuf + (size_t)N * HIDDEN;           // G*HIDDEN
    uintptr_t wp  = (uintptr_t)(z + (size_t)G * HIDDEN);
    wp = (wp + 63) & ~(uintptr_t)63;                     // 64B align for 32B vector loads
    _Float16* w1f = (_Float16*)wp;                       // IN_CH*HIDDEN f16, swizzled

    const int* esrc = ei;
    const int* edst = ei + E;

    zero_f32_kernel<<<2048, 256, 0, stream>>>(h2, zcount);
    cvt_w1_swizzle_kernel<<<(IN_CH * HIDDEN) / 256, 256, 0, stream>>>(w1, w1f);
    deg_kernel<<<(E + 255) / 256, 256, 0, stream>>>(edst, deg, E);
    dinv_kernel<<<(N + 255) / 256, 256, 0, stream>>>(deg, N);

    gemm_x_w1_kernel<<<N / 16, 256, 0, stream>>>(x, w1f, hbuf);

    long long aggThreads = (long long)(E + N) * 32;
    aggregate_kernel<<<(int)((aggThreads + 255) / 256), 256, 0, stream>>>(
        esrc, edst, deg, hbuf, h2, E, N);

    long long nq = (long long)N * (HIDDEN / 4);
    bias_relu_pool_kernel<<<(int)((nq + 255) / 256), 256, 0, stream>>>(
        h2, b1, batch, pooled, cnt, N);

    head1_kernel<<<G, HIDDEN, 0, stream>>>(pooled, cnt, w2, b2, z);
    head2_kernel<<<G, HIDDEN, 0, stream>>>(z, w3, b3, out);
}
